// MoELayer_2173253451902
// MI455X (gfx1250) — compile-verified, matrix operand order checked by reading, not run on hardware
//
#include <hip/hip_runtime.h>
#include <hip/hip_bf16.h>

#define N_TOK 8192
#define D_DIM 1024
#define E_NUM 8
#define F_DIM 4096
#define TILE_M 64
#define FB 64
#define KSLAB 64
#define DG 64
#define NSLAB (D_DIM / KSLAB)   // 16
#define NGRP  (D_DIM / DG)      // 16

#define XS_ST 1032   // Xs row stride (bf16 elems): 2064 B/row, 16B aligned
#define HS_ST 72
#define TW1_ST 72
#define TW2_ST 72

// dynamic LDS layout (bytes)
#define XS_OFF   0
#define HS_OFF   132096      // 64*1032*2
#define TWA_OFF  141312      // +64*72*2
#define TWB_OFF  150528      // +9216
#define RTOK_OFF 159744      // +9216
#define RW_OFF   160000
#define SMEM_BYTES 160256    // fits 160KB CU-mode LDS budget

typedef __bf16 bf16t;
typedef __attribute__((ext_vector_type(16))) bf16t v16bf;
typedef __attribute__((ext_vector_type(8)))  float v8f;

union Frag16 { v16bf v; uint4 u[2]; };

__device__ inline v8f zero8() {
  v8f z;
#pragma unroll
  for (int i = 0; i < 8; ++i) z[i] = 0.0f;
  return z;
}

__device__ inline float gelu_tanh(float x) {
  return 0.5f * x * (1.0f + tanhf(0.7978845608028654f * (x + 0.044715f * x * x * x)));
}

// ---- CDNA5 async copy: global -> LDS, 16B per lane, tracked by ASYNCcnt ----
__device__ __forceinline__ void async_copy_b128(unsigned lds_byte_off,
                                                unsigned gv_byte_off,
                                                const void* sbase) {
  asm volatile("global_load_async_to_lds_b128 %0, %1, %2"
               :: "v"(lds_byte_off), "v"(gv_byte_off), "s"(sbase)
               : "memory");
}
__device__ __forceinline__ void wait_async0() {
  asm volatile("s_wait_asynccnt 0x0" ::: "memory");
}
// in-order retirement: <=1 outstanding means everything except the most
// recent single-copy stage has landed
__device__ __forceinline__ void wait_async1() {
  asm volatile("s_wait_asynccnt 0x1" ::: "memory");
}

// ---------------- zero workspace ----------------
__global__ void zero_kernel(float* accbuf, int* counts, size_t n) {
  size_t i = (size_t)blockIdx.x * 256u + threadIdx.x;
  if (i < E_NUM) counts[i] = 0;
  size_t stride = (size_t)gridDim.x * 256u;
  for (; i < n; i += stride) accbuf[i] = 0.0f;
}

// ---------------- one-time weight convert + transpose to bf16 ----------------
// W1T[e][f][k] = w1[e][k][f]   (E,F,D)  -> B-fragments for X@W1 contiguous in k
// W2T[e][d][f] = w2[e][f][d]   (E,D,F)  -> B-fragments for H@W2 contiguous in f
__global__ void convert_w_kernel(const float* __restrict__ w1,
                                 const float* __restrict__ w2,
                                 bf16t* __restrict__ W1T,
                                 bf16t* __restrict__ W2T) {
  const size_t total = (size_t)E_NUM * F_DIM * D_DIM;
  const size_t stride = (size_t)gridDim.x * blockDim.x;
  for (size_t i = (size_t)blockIdx.x * blockDim.x + threadIdx.x; i < total; i += stride) {
    size_t e   = i >> 22;          // / (F*D)
    size_t rem = i & 4194303u;
    size_t f   = rem >> 10;        // / D
    size_t k   = rem & 1023u;
    W1T[i] = (bf16t)w1[((e << 10) + k) * F_DIM + f];
    size_t d   = rem >> 12;        // / F
    size_t f2  = rem & 4095u;
    W2T[i] = (bf16t)w2[((e << 12) + f2) * D_DIM + d];
  }
}

// ---------------- router: logits, softmax, top-2, bucket append, bf16 convert ----------------
__global__ void router_kernel(const float* __restrict__ hs,
                              const float* __restrict__ gw,
                              const float* __restrict__ gb,
                              bf16t* __restrict__ Xb,
                              int* __restrict__ counts,
                              int* __restrict__ etok,
                              float* __restrict__ ewt) {
  const int token = blockIdx.x;
  const int t = threadIdx.x;
  __shared__ float red[256];
  __shared__ float logit_sh[E_NUM];

  const float* x = hs + (size_t)token * D_DIM;
  float4 xv = *(const float4*)(x + 4 * t);

  union { bf16t b[4]; uint2 u; } cv;
  cv.b[0] = (bf16t)xv.x; cv.b[1] = (bf16t)xv.y;
  cv.b[2] = (bf16t)xv.z; cv.b[3] = (bf16t)xv.w;
  *(uint2*)(Xb + (size_t)token * D_DIM + 4 * t) = cv.u;

  float part[E_NUM];
#pragma unroll
  for (int e = 0; e < E_NUM; ++e) {
    const float* g = gw + (size_t)(4 * t) * E_NUM + e;
    part[e] = xv.x * g[0] + xv.y * g[E_NUM] + xv.z * g[2 * E_NUM] + xv.w * g[3 * E_NUM];
  }
  for (int e = 0; e < E_NUM; ++e) {
    red[t] = part[e];
    __syncthreads();
    for (int s = 128; s > 0; s >>= 1) {
      if (t < s) red[t] += red[t + s];
      __syncthreads();
    }
    if (t == 0) logit_sh[e] = red[0] + gb[e];
    __syncthreads();
  }

  if (t == 0) {
    float m = logit_sh[0];
#pragma unroll
    for (int e = 1; e < E_NUM; ++e) m = fmaxf(m, logit_sh[e]);
    float p[E_NUM], sum = 0.0f;
#pragma unroll
    for (int e = 0; e < E_NUM; ++e) { p[e] = expf(logit_sh[e] - m); sum += p[e]; }
    float inv = 1.0f / sum;
#pragma unroll
    for (int e = 0; e < E_NUM; ++e) p[e] *= inv;
    int i1 = 0; float g1 = p[0];
#pragma unroll
    for (int e = 1; e < E_NUM; ++e) if (p[e] > g1) { g1 = p[e]; i1 = e; }
    int i2 = (i1 == 0) ? 1 : 0; float g2 = p[i2];
#pragma unroll
    for (int e = 0; e < E_NUM; ++e) if (e != i1 && p[e] > g2) { g2 = p[e]; i2 = e; }

    int pos1 = atomicAdd(&counts[i1], 1);
    etok[(size_t)i1 * N_TOK + pos1] = token;
    ewt [(size_t)i1 * N_TOK + pos1] = g1;
    int pos2 = atomicAdd(&counts[i2], 1);
    etok[(size_t)i2 * N_TOK + pos2] = token;
    ewt [(size_t)i2 * N_TOK + pos2] = g2;
  }
}

// ---------------- expert FFN: double-buffered async staging + bf16 WMMA ----------------
// 512 threads = 16 wave32; 1 WMMA tile per wave per stage -> acc[16] = 128 VGPRs.
__global__ void __launch_bounds__(512, 1)
moe_ffn_kernel(const bf16t* __restrict__ W1T, const float* __restrict__ b1,
               const bf16t* __restrict__ W2T, const float* __restrict__ b2,
               const bf16t* __restrict__ Xb,
               const int* __restrict__ counts,
               const int* __restrict__ etok,
               const float* __restrict__ ewt,
               float* __restrict__ accbuf) {
  const int e = blockIdx.y;
  const int tile = blockIdx.x;
  const int t = threadIdx.x;
  const int wave = t >> 5;
  const int lane = t & 31;

  const int count = counts[e];
  if (tile * TILE_M >= count) return;

  extern __shared__ char smem[];
  bf16t* Xs     = (bf16t*)(smem + XS_OFF);     // [64][1032]
  bf16t* Hs     = (bf16t*)(smem + HS_OFF);     // [64][72]
  bf16t* TWbuf[2];
  TWbuf[0] = (bf16t*)(smem + TWA_OFF);         // [64][72] staging (double-buffered)
  TWbuf[1] = (bf16t*)(smem + TWB_OFF);
  int*   rowTok = (int*)  (smem + RTOK_OFF);
  float* rowW   = (float*)(smem + RW_OFF);

  const unsigned lds_base = __builtin_amdgcn_groupstaticsize();
  const unsigned xs_lds = lds_base + XS_OFF;
  const unsigned tw_lds[2] = { lds_base + TWA_OFF, lds_base + TWB_OFF };

  if (t < TILE_M) {
    int idx = tile * TILE_M + t;
    bool ok = idx < count;
    int src = ok ? idx : 0;
    rowTok[t] = etok[(size_t)e * N_TOK + src];
    rowW[t]   = ok ? ewt[(size_t)e * N_TOK + src] : 0.0f;
  }
  __syncthreads();

  // ---- gather X tile (bf16) via async copies; completion folded into the
  //      first pipeline wait (in-order ASYNCcnt retirement) ----
  for (int i = 0; i < 16; ++i) {
    int chunk = i * 512 + t;            // 8192 chunks of 8 bf16 (16B)
    int r = chunk >> 7;
    int c = (chunk & 127) << 3;
    unsigned go = (unsigned)((rowTok[r] * D_DIM + c) * 2);
    async_copy_b128(xs_lds + (unsigned)(r * XS_ST + c) * 2, go, Xb);
  }

  const bf16t* W1Te = W1T + (size_t)e * F_DIM * D_DIM;
  const bf16t* W2Te = W2T + (size_t)e * D_DIM * F_DIM;

  v8f acc[NGRP];
#pragma unroll
  for (int g = 0; g < NGRP; ++g) acc[g] = zero8();

  // unified wave->tile mapping for both phases: 4x4 tile grid per stage
  const int mi   = wave & 3;        // M tile (rows mi*16..+15)
  const int nt   = wave >> 2;       // N tile within 64-wide stage (0..3)
  const int lrow = lane & 15;
  const int kb8  = (lane >> 4) << 3;

  // per-thread staging coords: one 16B chunk per thread per stage (512*8=4096 elems)
  const int srow = t >> 3;          // 0..63 (f for TW1, d for TW2)
  const int scol = (t & 7) << 3;    // 0..56 step 8

  for (int fc = 0; fc < F_DIM / FB; ++fc) {
    const int fcol = fc * FB;
    v8f hacc = zero8();

    // ======== phase 1: h = X @ W1[:, fcol:fcol+64] ========
    // prologue: stage slab 0 into buffer 0 (prev readers fenced by last barrier)
    async_copy_b128(tw_lds[0] + (unsigned)(srow * TW1_ST + scol) * 2,
                    (unsigned)(((fcol + srow) * D_DIM + scol) * 2), W1Te);
    for (int s = 0; s < NSLAB; ++s) {
      const int ks = s * KSLAB;
      if (s + 1 < NSLAB) {
        async_copy_b128(tw_lds[(s + 1) & 1] + (unsigned)(srow * TW1_ST + scol) * 2,
                        (unsigned)(((fcol + srow) * D_DIM + ks + KSLAB + scol) * 2), W1Te);
        wait_async1();   // slab s resident; slab s+1 still in flight
      } else {
        wait_async0();
      }
      __syncthreads();
      const bf16t* tw = TWbuf[s & 1];
#pragma unroll
      for (int kk = 0; kk < KSLAB; kk += 32) {
        Frag16 a, b;
        const bf16t* ap = Xs + (mi * 16 + lrow) * XS_ST + (ks + kk + kb8);
        a.u[0] = *(const uint4*)(ap);
        a.u[1] = *(const uint4*)(ap + 16);
        const bf16t* bp = tw + (nt * 16 + lrow) * TW1_ST + (kk + kb8);
        b.u[0] = *(const uint4*)(bp);
        b.u[1] = *(const uint4*)(bp + 16);
        hacc = __builtin_amdgcn_wmma_f32_16x16x32_bf16(
            false, a.v, false, b.v, (short)0, hacc, false, false);
      }
      __syncthreads();   // readers done before anyone restages this buffer
    }

    // ======== phase 2 prologue: stage group 0 under the gelu VALU work ========
    async_copy_b128(tw_lds[0] + (unsigned)(srow * TW2_ST + scol) * 2,
                    (unsigned)((srow * F_DIM + fcol + scol) * 2), W2Te);

    // ---- bias + gelu -> Hs (bf16): one 16x16 tile per wave ----
    {
      int coln = nt * 16 + lrow;
      float bb = b1[(size_t)e * F_DIM + fcol + coln];
#pragma unroll
      for (int v = 0; v < 8; ++v) {
        float xv = hacc[v] + bb;
        Hs[(mi * 16 + kb8 + v) * HS_ST + coln] = (bf16t)gelu_tanh(xv);
      }
    }

    // ======== phase 2: Y += h @ W2[fcol:fcol+64, :] over 16 x 64-col groups ========
#pragma unroll
    for (int g = 0; g < NGRP; ++g) {
      if (g + 1 < NGRP) {
        async_copy_b128(tw_lds[(g + 1) & 1] + (unsigned)(srow * TW2_ST + scol) * 2,
                        (unsigned)((((g + 1) * DG + srow) * F_DIM + fcol + scol) * 2), W2Te);
        wait_async1();   // group g resident
      } else {
        wait_async0();
      }
      __syncthreads();   // also publishes the Hs stores on first iteration
      const bf16t* tw = TWbuf[g & 1];
#pragma unroll
      for (int kk = 0; kk < FB; kk += 32) {
        Frag16 a, b;
        const bf16t* ap = Hs + (mi * 16 + lrow) * HS_ST + (kk + kb8);
        a.u[0] = *(const uint4*)(ap);
        a.u[1] = *(const uint4*)(ap + 16);
        const bf16t* bp = tw + (nt * 16 + lrow) * TW2_ST + (kk + kb8);
        b.u[0] = *(const uint4*)(bp);
        b.u[1] = *(const uint4*)(bp + 16);
        acc[g] = __builtin_amdgcn_wmma_f32_16x16x32_bf16(
            false, a.v, false, b.v, (short)0, acc[g], false, false);
      }
      __syncthreads();
    }
  }

  // ---- epilogue: +b2, scale by gate, scatter-accumulate ----
#pragma unroll
  for (int g = 0; g < NGRP; ++g) {
    int col = g * DG + nt * 16 + lrow;
    float bb = b2[(size_t)e * D_DIM + col];
#pragma unroll
    for (int v = 0; v < 8; ++v) {
      int r = mi * 16 + kb8 + v;
      float y = acc[g][v] + bb;
      atomicAdd(accbuf + (size_t)rowTok[r] * D_DIM + col, rowW[r] * y);
    }
  }
}

// ---------------- residual + layernorm ----------------
__global__ void ln_kernel(const float* __restrict__ accbuf,
                          const float* __restrict__ inp,
                          const float* __restrict__ gamma,
                          const float* __restrict__ beta,
                          float* __restrict__ out) {
  const int token = blockIdx.x;
  const int t = threadIdx.x;
  __shared__ float red[256];
  __shared__ float mu_sh, rstd_sh;

  float4 a = *(const float4*)(accbuf + (size_t)token * D_DIM + 4 * t);
  float4 b = *(const float4*)(inp    + (size_t)token * D_DIM + 4 * t);
  float s0 = a.x + b.x, s1 = a.y + b.y, s2 = a.z + b.z, s3 = a.w + b.w;

  red[t] = s0 + s1 + s2 + s3;
  __syncthreads();
  for (int s = 128; s > 0; s >>= 1) {
    if (t < s) red[t] += red[t + s];
    __syncthreads();
  }
  if (t == 0) mu_sh = red[0] * (1.0f / D_DIM);
  __syncthreads();
  float mu = mu_sh;
  float d0 = s0 - mu, d1 = s1 - mu, d2 = s2 - mu, d3 = s3 - mu;
  red[t] = d0 * d0 + d1 * d1 + d2 * d2 + d3 * d3;
  __syncthreads();
  for (int s = 128; s > 0; s >>= 1) {
    if (t < s) red[t] += red[t + s];
    __syncthreads();
  }
  if (t == 0) rstd_sh = rsqrtf(red[0] * (1.0f / D_DIM) + 1e-5f);
  __syncthreads();
  float rstd = rstd_sh;

  int c = 4 * t;
  out[(size_t)token * D_DIM + c + 0] = d0 * rstd * gamma[c + 0] + beta[c + 0];
  out[(size_t)token * D_DIM + c + 1] = d1 * rstd * gamma[c + 1] + beta[c + 1];
  out[(size_t)token * D_DIM + c + 2] = d2 * rstd * gamma[c + 2] + beta[c + 2];
  out[(size_t)token * D_DIM + c + 3] = d3 * rstd * gamma[c + 3] + beta[c + 3];
}

extern "C" void kernel_launch(void* const* d_in, const int* in_sizes, int n_in,
                              void* d_out, int out_size, void* d_ws, size_t ws_size,
                              hipStream_t stream) {
  (void)in_sizes; (void)n_in; (void)out_size; (void)ws_size;
  const float* hs    = (const float*)d_in[0];
  const float* inp   = (const float*)d_in[1];
  const float* gw    = (const float*)d_in[2];
  const float* gb    = (const float*)d_in[3];
  const float* w1    = (const float*)d_in[4];
  const float* b1    = (const float*)d_in[5];
  const float* w2    = (const float*)d_in[6];
  const float* b2    = (const float*)d_in[7];
  const float* gamma = (const float*)d_in[8];
  const float* beta  = (const float*)d_in[9];
  float* out = (float*)d_out;

  char* ws = (char*)d_ws;
  size_t off = 0;
  int*   counts = (int*)(ws + off);   off += 256;
  int*   etok   = (int*)(ws + off);   off += (size_t)E_NUM * N_TOK * sizeof(int);
  float* ewt    = (float*)(ws + off); off += (size_t)E_NUM * N_TOK * sizeof(float);
  bf16t* Xb     = (bf16t*)(ws + off); off += (size_t)N_TOK * D_DIM * sizeof(bf16t);
  float* accb   = (float*)(ws + off); off += (size_t)N_TOK * D_DIM * sizeof(float);
  bf16t* W1T    = (bf16t*)(ws + off); off += (size_t)E_NUM * F_DIM * D_DIM * sizeof(bf16t);
  bf16t* W2T    = (bf16t*)(ws + off); off += (size_t)E_NUM * F_DIM * D_DIM * sizeof(bf16t);

  zero_kernel<<<dim3(2048), dim3(256), 0, stream>>>(accb, counts, (size_t)N_TOK * D_DIM);
  convert_w_kernel<<<dim3(8192), dim3(256), 0, stream>>>(w1, w2, W1T, W2T);
  router_kernel<<<dim3(N_TOK), dim3(256), 0, stream>>>(hs, gw, gb, Xb, counts, etok, ewt);
  moe_ffn_kernel<<<dim3(N_TOK / TILE_M, E_NUM), dim3(512), SMEM_BYTES, stream>>>(
      W1T, b1, W2T, b2, Xb, counts, etok, ewt, accb);
  ln_kernel<<<dim3(N_TOK), dim3(256), 0, stream>>>(accb, inp, gamma, beta, out);
}
